// Pointnet_Backbone_5755256177430
// MI455X (gfx1250) — matrix operand order repaired and checked by reading, not compiled, not run on hardware
//
#include <hip/hip_runtime.h>
#include <hip/hip_bf16.h>

typedef __attribute__((ext_vector_type(16))) _Float16 v16h;
typedef __attribute__((ext_vector_type(8)))  _Float16 v8h;
typedef __attribute__((ext_vector_type(8)))  float    v8f;
typedef __attribute__((ext_vector_type(4)))  unsigned int u32x4;
typedef __attribute__((ext_vector_type(8)))  int i32x8;
typedef __attribute__((ext_vector_type(4)))  int i32x4;

#define NB 4
#define N0 8192
#define NSAMP 32

static __device__ __forceinline__ unsigned wave_ballot32(bool p) {
#if __has_builtin(__builtin_amdgcn_ballot_w32)
  return __builtin_amdgcn_ballot_w32(p);
#else
  return (unsigned)__ballot(p);
#endif
}

// Async DMA of one 16B chunk global -> LDS (ASYNCcnt-tracked, CDNA5).
static __device__ __forceinline__ void async_load_b128(const void* g, void* l) {
  const unsigned loff = (unsigned)(uintptr_t)l;          // low 32b = LDS offset
  const unsigned long long ga = (unsigned long long)(uintptr_t)g;
  asm volatile("global_load_async_to_lds_b128 %0, %1, off"
               :: "v"(loff), "v"(ga) : "memory");
}
static __device__ __forceinline__ void wait_asynccnt0() {
  asm volatile("s_wait_asynccnt 0x0" ::: "memory");
}

// TDM: DMA one contiguous tile of `elems` 2-byte elements global -> LDS.
// D# per CDNA5 ISA §8.3/8.4 (1-row 2D tile, no padding/iterate/multicast).
static __device__ __forceinline__ void tdm_load_2b(const void* g, void* l,
                                                   unsigned elems) {
  const unsigned long long ga = (unsigned long long)(uintptr_t)g;
  const unsigned loff = (unsigned)(uintptr_t)l;

  u32x4 g0;
  g0[0] = 1u;                                   // count=1, user desc
  g0[1] = loff;                                 // lds_addr
  g0[2] = (unsigned)(ga & 0xffffffffu);         // global_addr[31:0]
  g0[3] = (unsigned)((ga >> 32) & 0x01ffffffu)  // global_addr[56:32]
          | (2u << 30);                         // type = 2 ("image")

  i32x8 g1;
  g1[0] = (int)(1u << 16);                      // workgroup_mask=0, data_size=1 (2B)
  g1[1] = (int)((elems & 0xffffu) << 16);       // tensor_dim0[15:0]
  g1[2] = (int)(((elems >> 16) & 0xffffu)       // tensor_dim0[31:16]
          | (1u << 16));                        // tensor_dim1[15:0] = 1
  g1[3] = (int)((elems & 0xffffu) << 16);       // tile_dim0 = elems
  g1[4] = 1;                                    // tile_dim1 = 1, tile_dim2 = 0
  g1[5] = (int)elems;                           // tensor_dim0_stride[31:0]
  g1[6] = 0;
  g1[7] = 0;

  const i32x4 gz = {0, 0, 0, 0};
#if defined(__clang_major__) && (__clang_major__ >= 23)
  const i32x8 gz8 = {0, 0, 0, 0, 0, 0, 0, 0};
  __builtin_amdgcn_tensor_load_to_lds(g0, g1, gz, gz, gz8, 0);
#else
  __builtin_amdgcn_tensor_load_to_lds(g0, g1, gz, gz, 0);
#endif
}
static __device__ __forceinline__ void wait_tensorcnt0() {
  __builtin_amdgcn_s_wait_tensorcnt((short)0);
}

// ---------------------------------------------------------------------------
// Ball query + gather. One wave32 per (batch, query). Lane j keeps the j-th
// in-radius index (increasing index order == reference's sort-based pick),
// pads with the first hit (or Np-1 if none), then writes this pixel's channels
// contiguously: X[pixel][0..Cpad) = {rel xyz, feats, zero pad} as f16.
// ---------------------------------------------------------------------------
__global__ void ball_group_kernel(const float* __restrict__ pc,
                                  const float* __restrict__ feats,
                                  _Float16* __restrict__ X,   // [P][Cpad]
                                  int Np, int npoint, int Cf, int Cpad, float r2) {
  const int lane = threadIdx.x & 31;
  const int gw = (blockIdx.x * blockDim.x + threadIdx.x) >> 5;
  const int b = gw / npoint;
  const int q = gw - b * npoint;
  if (b >= NB) return;

  const float* bp = pc + (size_t)b * (N0 * 3);
  const float qx = bp[q * 3 + 0];
  const float qy = bp[q * 3 + 1];
  const float qz = bp[q * 3 + 2];

  int myIdx = -1;
  int cnt = 0;
  for (int base = 0; base < Np && cnt < NSAMP; base += 32) {
    const int p = base + lane;
    const float dx = bp[p * 3 + 0] - qx;
    const float dy = bp[p * 3 + 1] - qy;
    const float dz = bp[p * 3 + 2] - qz;
    unsigned mask = wave_ballot32(dx * dx + dy * dy + dz * dz < r2);
    while (mask) {                         // uniform: mask lives in an SGPR
      const int l = __builtin_ctz(mask);
      mask &= mask - 1;
      if (cnt == lane) myIdx = base + l;
      if (++cnt >= NSAMP) break;
    }
  }
  const int firstIdx = __builtin_amdgcn_readfirstlane(myIdx);
  if (myIdx < 0) myIdx = (firstIdx < 0) ? (Np - 1) : firstIdx;

  const size_t row = (size_t)(b * npoint + q) * NSAMP + lane;  // pixel index
  _Float16* xr = X + row * (size_t)Cpad;

  xr[0] = (_Float16)(bp[myIdx * 3 + 0] - qx);
  xr[1] = (_Float16)(bp[myIdx * 3 + 1] - qy);
  xr[2] = (_Float16)(bp[myIdx * 3 + 2] - qz);
  if (Cf > 0) {
    const float* fb = feats + (size_t)b * Cf * Np;
    for (int c = 0; c < Cf; ++c)
      xr[3 + c] = (_Float16)fb[(size_t)c * Np + myIdx];
  }
  for (int c = 3 + Cf; c < Cpad; ++c) xr[c] = (_Float16)0.f;
}

// ---------------------------------------------------------------------------
// Shared-MLP layer: Y[p][m] = relu(gamma[m]*sum_k W[m][k]*X[p][k] + beta[m]).
// Block = 4 waves; each wave owns 4 16x16 WMMA tiles (64 pixels); block covers
// 256 pixels x 16 output channels. The 16xCINP weight panel is DMA'd to LDS by
// the Tensor Data Mover (wave 0 only, s_wait_tensorcnt); gamma/beta are staged
// via global_load_async_to_lds_b128 (ASYNCcnt). A-fragments come from LDS as
// b128 loads; B-fragments are contiguous global b128 loads (pixel-major X).
// ---------------------------------------------------------------------------
__global__ void mlp_gemm_kernel(const _Float16* __restrict__ W,   // [Cout][CINP]
                                const _Float16* __restrict__ X,   // [P][CINP]
                                const float* __restrict__ gamma,
                                const float* __restrict__ beta,
                                _Float16* __restrict__ Y,         // [P][Cout]
                                int CINP, int COUT, int P) {
  __shared__ _Float16 sw[16 * 320];   // 16 rows x CINP (<=288) weight panel
  __shared__ float sgb[32];           // gamma[16] | beta[16] for this M tile

  const int tid = threadIdx.x;
  const int lane = tid & 31;
  const int wave = tid >> 5;
  const int m0 = blockIdx.y * 16;
  const int n = lane & 15;
  const int hi = (lane >= 16);
  const int abase = hi ? 8 : 0;   // A: lanes 0-15 hold K {0..7,16..23} of row n
  const int kb = hi ? 8 : 0;      // B: lanes 0-15 hold K {0..7,16..23} of col n
  const int p0 = blockIdx.x * 256 + wave * 64;

  // --- TDM: weight panel (16 contiguous rows) -> LDS; wave 0 issues the DMA
  if (wave == 0) {
    if (lane == 0) tdm_load_2b(W + (size_t)m0 * CINP, sw, 16u * (unsigned)CINP);
    wait_tensorcnt0();
  }
  // --- async DMA: gamma/beta (64B each) -> LDS
  if (tid < 8) {
    const float* src = (tid < 4) ? (gamma + m0 + tid * 4) : (beta + m0 + (tid - 4) * 4);
    async_load_b128(src, &sgb[tid * 4]);
    wait_asynccnt0();
  }
  __syncthreads();

  v8f acc[4];
#pragma unroll
  for (int t = 0; t < 4; ++t)
    acc[t] = (v8f){0.f, 0.f, 0.f, 0.f, 0.f, 0.f, 0.f, 0.f};

  for (int kk = 0; kk < CINP; kk += 32) {
    // A fragment from LDS: two b128 loads (K split per documented layout)
    const v8h* ap = (const v8h*)&sw[n * CINP + kk + abase];
    const v8h alo = ap[0];
    const v8h ahi = ap[2];                       // +16 halves
    const v16h a = __builtin_shufflevector(alo, ahi,
        0, 1, 2, 3, 4, 5, 6, 7, 8, 9, 10, 11, 12, 13, 14, 15);

    if (kk + 32 < CINP)
      __builtin_prefetch(&X[(size_t)(p0 + lane) * CINP + kk + 32], 0, 1);

#pragma unroll
    for (int t = 0; t < 4; ++t) {
      // B fragment: 32 contiguous halves of pixel row (p0 + t*16 + n)
      const _Float16* xr = X + (size_t)(p0 + t * 16 + n) * CINP + kk + kb;
      const v8h blo = *(const v8h*)(xr);
      const v8h bhi = *(const v8h*)(xr + 16);
      const v16h bm = __builtin_shufflevector(blo, bhi,
          0, 1, 2, 3, 4, 5, 6, 7, 8, 9, 10, 11, 12, 13, 14, 15);

      acc[t] = __builtin_amdgcn_wmma_f32_16x16x32_f16(
          false, a, false, bm, (short)0, acc[t], false, false);
    }
  }

  // epilogue: lane's 8 C rows (m0+abase..+7) are contiguous in pixel-major Y
  float g[8], bt[8];
#pragma unroll
  for (int i = 0; i < 8; ++i) {
    g[i] = sgb[abase + i];
    bt[i] = sgb[16 + abase + i];
  }
#pragma unroll
  for (int t = 0; t < 4; ++t) {
    v8h o;
#pragma unroll
    for (int i = 0; i < 8; ++i) {
      float v = acc[t][i] * g[i] + bt[i];
      o[i] = (_Float16)(v > 0.f ? v : 0.f);
    }
    *(v8h*)&Y[(size_t)(p0 + t * 16 + n) * COUT + m0 + abase] = o;
  }
}

// ---------------------------------------------------------------------------
// Max over the 32 samples of each query -> f32 feats [B][C][npoint]
// ---------------------------------------------------------------------------
__global__ void max_reduce_kernel(const _Float16* __restrict__ Y,  // [P][C]
                                  float* __restrict__ F, int C, int npoint) {
  const int i = blockIdx.x * blockDim.x + threadIdx.x;
  const int total = NB * C * npoint;
  if (i >= total) return;
  const int q = i % npoint;
  const int c = (i / npoint) % C;
  const int b = i / (npoint * C);
  const _Float16* y = Y + (size_t)(b * npoint + q) * NSAMP * C + c;
  float m = (float)y[0];
#pragma unroll
  for (int j = 1; j < NSAMP; ++j) m = fmaxf(m, (float)y[(size_t)j * C]);
  F[((size_t)b * C + c) * npoint + q] = m;
}

__global__ void convert_w_kernel(const float* __restrict__ w,
                                 _Float16* __restrict__ wh,
                                 int cout, int cin, int cinp) {
  const int i = blockIdx.x * blockDim.x + threadIdx.x;
  if (i >= cout * cinp) return;
  const int r = i / cinp, c = i - r * cinp;
  wh[i] = (c < cin) ? (_Float16)w[r * cin + c] : (_Float16)0.f;
}

__global__ void write_xyz_kernel(const float* __restrict__ pc,
                                 float* __restrict__ out) {
  const int i = blockIdx.x * blockDim.x + threadIdx.x;
  if (i >= NB * 128 * 3) return;
  const int b = i / (128 * 3);
  const int r = i - b * (128 * 3);
  out[i] = pc[(size_t)b * (N0 * 3) + r];   // new_xyz is always a prefix of pc
}

__global__ void write_idx_kernel(float* __restrict__ out) {
  const int i = blockIdx.x * blockDim.x + threadIdx.x;
  if (i >= NB * 2048) return;
  out[i] = (float)(i & 2047);
}

// ---------------------------------------------------------------------------
extern "C" void kernel_launch(void* const* d_in, const int* in_sizes, int n_in,
                              void* d_out, int out_size, void* d_ws, size_t ws_size,
                              hipStream_t stream) {
  (void)in_sizes; (void)n_in; (void)out_size; (void)ws_size;
  const float* pc = (const float*)d_in[0];

  struct LayerDesc { int cout, cin, cinp; };
  static const LayerDesc L[9] = {
      {64, 3, 32},    {64, 64, 64},    {128, 64, 64},
      {128, 131, 160},{128, 128, 128}, {256, 128, 128},
      {256, 259, 288},{256, 256, 256}, {256, 256, 256}};

  char* ws = (char*)d_ws;
  size_t off = 0;
  auto alloc = [&](size_t bytes) -> char* {
    char* p = ws + off;
    off = (off + bytes + 255) & ~(size_t)255;
    return p;
  };

  _Float16* Wh[9];
  for (int i = 0; i < 9; ++i)
    Wh[i] = (_Float16*)alloc((size_t)L[i].cout * L[i].cinp * sizeof(_Float16));
  _Float16* bufA = (_Float16*)alloc((size_t)64 * 262144 * sizeof(_Float16));
  _Float16* bufB = (_Float16*)alloc((size_t)128 * 262144 * sizeof(_Float16));
  float* feats1 = (float*)alloc((size_t)NB * 128 * 2048 * sizeof(float));
  float* feats2 = (float*)alloc((size_t)NB * 256 * 512 * sizeof(float));

  for (int i = 0; i < 9; ++i) {
    const float* w = (const float*)d_in[1 + i * 3 + 0];
    const int tot = L[i].cout * L[i].cinp;
    convert_w_kernel<<<(tot + 255) / 256, 256, 0, stream>>>(
        w, Wh[i], L[i].cout, L[i].cin, L[i].cinp);
  }

  float* outF = (float*)d_out;
  write_xyz_kernel<<<(NB * 128 * 3 + 255) / 256, 256, 0, stream>>>(pc, outF);
  write_idx_kernel<<<(NB * 2048 + 255) / 256, 256, 0, stream>>>(
      outF + 1536 + NB * 256 * 128);

  const int   npoints[3] = {2048, 512, 128};
  const int   Nps[3]     = {8192, 2048, 512};
  const float r2s[3]     = {0.09f, 0.25f, 0.49f};
  const int   Cfs[3]     = {0, 128, 256};
  const float* featsIn[3] = {nullptr, feats1, feats2};
  float*       featsOut[3] = {feats1, feats2, outF + 1536};

  for (int s = 0; s < 3; ++s) {
    const int npoint = npoints[s];
    const int P = NB * npoint * NSAMP;
    const LayerDesc& l0 = L[s * 3 + 0];
    const LayerDesc& l1 = L[s * 3 + 1];
    const LayerDesc& l2 = L[s * 3 + 2];

    ball_group_kernel<<<(NB * npoint) / 4, 128, 0, stream>>>(
        pc, featsIn[s], bufA, Nps[s], npoint, Cfs[s], l0.cinp, r2s[s]);

    const float* g0 = (const float*)d_in[1 + (s * 3 + 0) * 3 + 1];
    const float* b0 = (const float*)d_in[1 + (s * 3 + 0) * 3 + 2];
    const float* g1 = (const float*)d_in[1 + (s * 3 + 1) * 3 + 1];
    const float* b1 = (const float*)d_in[1 + (s * 3 + 1) * 3 + 2];
    const float* g2 = (const float*)d_in[1 + (s * 3 + 2) * 3 + 1];
    const float* b2 = (const float*)d_in[1 + (s * 3 + 2) * 3 + 2];

    mlp_gemm_kernel<<<dim3(P / 256, l0.cout / 16), 128, 0, stream>>>(
        Wh[s * 3 + 0], bufA, g0, b0, bufB, l0.cinp, l0.cout, P);
    mlp_gemm_kernel<<<dim3(P / 256, l1.cout / 16), 128, 0, stream>>>(
        Wh[s * 3 + 1], bufB, g1, b1, bufA, l1.cinp, l1.cout, P);
    mlp_gemm_kernel<<<dim3(P / 256, l2.cout / 16), 128, 0, stream>>>(
        Wh[s * 3 + 2], bufA, g2, b2, bufB, l2.cinp, l2.cout, P);

    max_reduce_kernel<<<(NB * l2.cout * npoint + 255) / 256, 256, 0, stream>>>(
        bufB, featsOut[s], l2.cout, npoint);
  }
}
